// NODEModel_35459249996655
// MI455X (gfx1250) — matrix operand order repaired and checked by reading, not compiled, not run on hardware
//
#include <hip/hip_runtime.h>
#include <math.h>

typedef _Float16 v16h __attribute__((ext_vector_type(16)));
typedef _Float16 v8h  __attribute__((ext_vector_type(8)));
typedef _Float16 v2h  __attribute__((ext_vector_type(2)));
typedef float    v8f  __attribute__((ext_vector_type(8)));

#define WIDTH 128
#define SOFTENING 0.1f

// gfx1250 has a hardware tanh transcendental (V_TANH_F32); use it if exposed.
__device__ __forceinline__ float fast_tanh(float x) {
#if __has_builtin(__builtin_amdgcn_tanhf)
    return __builtin_amdgcn_tanhf(x);
#elif __has_builtin(__builtin_amdgcn_tanh_f32)
    return __builtin_amdgcn_tanh_f32(x);
#else
    return tanhf(x);
#endif
}

// LDS layout (halves unless noted):
//  WtF : [3][128][128]  forward B operand  (WtF[k][n][c] = Wh[k][c][n])
//  Wb  : [3][128][128]  backward B operand (row-major copy of Wh)
//  WinS: [4][128]       padded input weights
//  stage: [8 waves][16*128] activation staging (D-layout <-> A-layout)
//  floats: binS[128], bhS[3*128], WoutS[128]
#define SM_WTF   0
#define SM_WB    (3*128*128)
#define SM_WIN   (SM_WB + 3*128*128)
#define SM_STAGE (SM_WIN + 512)
#define SM_FLT_B (SM_STAGE + 8*2048)                 // in halves
#define SMEM_BYTES (SM_FLT_B*2 + (128 + 3*128 + 128)*4)

// ---------------------------------------------------------------------------
// Kernel 1: build spherical coords + quadrature-point MLP inputs
// ---------------------------------------------------------------------------
__global__ void prep_kernel(const float4* __restrict__ tx,
                            float* __restrict__ ic_in,   // N x 4
                            float* __restrict__ dp_in,   // 3N x 4
                            int N)
{
    int i = blockIdx.x * blockDim.x + threadIdx.x;
    if (i >= N) return;
    float4 row = tx[i];
    float t = row.x, x = row.y, y = row.z, z = row.w;
    float r = sqrtf(x*x + y*y + z*z + 1e-12f);
    float c = fminf(fmaxf(z / r, -1.f), 1.f);
    float th = acosf(c);
    float ph = atan2f(y, x);
    ic_in[i*4+0] = r; ic_in[i*4+1] = th; ic_in[i*4+2] = ph; ic_in[i*4+3] = 0.f;
    const float nodes[3] = {-0.7745966692f, 0.f, 0.7745966692f};
    float a = 0.5f * t;     // (t - T0)/2 with T0 = 0; b == a
    #pragma unroll
    for (int q = 0; q < 3; ++q) {
        float ts = a * nodes[q] + a;
        int o = (3*i + q) * 4;
        dp_in[o+0] = ts; dp_in[o+1] = r; dp_in[o+2] = th; dp_in[o+3] = ph;
    }
}

// ---------------------------------------------------------------------------
// Kernel 2: batched MLP forward + input-gradient via WMMA
//   y(row) = Wout . h3 + bout,  h_{k} = tanh(h_{k-1} Wh[k-1] + bh),
//   h0 = tanh(x Win + bin).  g_out = d y / d x  (4 components).
// ---------------------------------------------------------------------------
__global__ __launch_bounds__(256, 1)
void mlp_eval_kernel(const float* __restrict__ in, int rows, int din,
                     const float* __restrict__ Win,  const float* __restrict__ bin,
                     const float* __restrict__ Wh,   const float* __restrict__ bh,
                     const float* __restrict__ Wout, const float* __restrict__ bout,
                     float* __restrict__ y_out, float* __restrict__ g_out)
{
    extern __shared__ char smem_raw[];
    _Float16* smh  = (_Float16*)smem_raw;
    _Float16* WtF  = smh + SM_WTF;
    _Float16* Wb   = smh + SM_WB;
    _Float16* WinS = smh + SM_WIN;
    _Float16* stg  = smh + SM_STAGE;
    float* binS  = (float*)(smh + SM_FLT_B);
    float* bhS   = binS + 128;
    float* WoutS = bhS + 3*128;

    const int tid = threadIdx.x;

    // ---- stage weights into LDS (fp32 -> fp16, plus transpose copy) ----
    for (int idx = tid; idx < 3*128*128; idx += 256) {
        float w = Wh[idx];
        int k = idx >> 14, rem = idx & 16383, c = rem >> 7, n = rem & 127;
        Wb[idx] = (_Float16)w;                       // row-major  (backward B)
        WtF[(k << 14) + (n << 7) + c] = (_Float16)w; // N-major (forward B)
    }
    for (int idx = tid; idx < 512; idx += 256) {
        int j = idx >> 7, n = idx & 127;
        WinS[idx] = (j < din) ? (_Float16)Win[j*128 + n] : (_Float16)0.f;
    }
    if (tid < 128) {
        binS[tid]  = bin[tid];
        WoutS[tid] = Wout[tid];
        bhS[tid]       = bh[tid];
        bhS[128 + tid] = bh[128 + tid];
        bhS[256 + tid] = bh[256 + tid];
    }
    __syncthreads();
    const float bout0 = bout[0];

    const int wave  = tid >> 5;
    const int lane  = tid & 31;
    const int lhalf = lane >> 4;    // 0 or 1
    const int lcol  = lane & 15;    // N within tile; also M for A-frag row
    _Float16* st = stg + wave * 2048;
    const float4* in4 = (const float4*)in;

    const int ntiles = rows >> 4;
    for (int tile = blockIdx.x * 8 + wave; tile < ntiles; tile += gridDim.x * 8) {
        const int r0 = tile << 4;

        // prefetch next tile's input rows while this tile computes
        {
            int nt = tile + gridDim.x * 8;
            if (nt < ntiles)
                __builtin_prefetch((const void*)&in4[(nt << 4) + lhalf*8], 0, 0);
        }

        v8f acc[8];
        v2h sp[4][8][4];   // (1 - h^2) per layer, packed f16, D-layout

        // ---------------- layer 0 (Din<=4): VALU ----------------
        float4 xin[8];
        #pragma unroll
        for (int v = 0; v < 8; ++v) xin[v] = in4[r0 + lhalf*8 + v];
        #pragma unroll
        for (int tn = 0; tn < 8; ++tn) {
            int Nn = tn*16 + lcol;
            float w0 = (float)WinS[Nn],       w1 = (float)WinS[128 + Nn];
            float w2 = (float)WinS[256 + Nn], w3 = (float)WinS[384 + Nn];
            float bv = binS[Nn];
            #pragma unroll
            for (int v = 0; v < 8; ++v) {
                float z = bv + xin[v].x*w0 + xin[v].y*w1 + xin[v].z*w2 + xin[v].w*w3;
                float h = fast_tanh(z);
                sp[0][tn][v >> 1][v & 1] = (_Float16)(1.f - h*h);
                st[(v + lhalf*8)*128 + Nn] = (_Float16)h;
            }
        }

        // ---------------- hidden layers: WMMA ----------------
        #pragma unroll
        for (int kk = 0; kk < 3; ++kk) {
            v16h afr[4];
            #pragma unroll
            for (int kf = 0; kf < 4; ++kf) {        // A frag: h_{kk} from staging
                int base = lcol*128 + kf*32 + lhalf*16;
                v8h lo = *(const v8h*)(st + base);
                v8h hi = *(const v8h*)(st + base + 8);
                #pragma unroll
                for (int e = 0; e < 8; ++e) { afr[kf][e] = lo[e]; afr[kf][8+e] = hi[e]; }
            }
            #pragma unroll
            for (int tn = 0; tn < 8; ++tn) {
                int Nn = tn*16 + lcol;
                float bv = bhS[kk*128 + Nn];
                v8f c;
                #pragma unroll
                for (int v = 0; v < 8; ++v) c[v] = bv;
                #pragma unroll
                for (int kf = 0; kf < 4; ++kf) {
                    v16h bfr = *(const v16h*)(WtF + kk*16384 + Nn*128 + kf*32 + lhalf*16);
                    c = __builtin_amdgcn_wmma_f32_16x16x32_f16(
                            false, afr[kf], false, bfr, (short)0, c, false, false);
                }
                acc[tn] = c;
            }
            #pragma unroll
            for (int tn = 0; tn < 8; ++tn) {
                int Nn = tn*16 + lcol;
                #pragma unroll
                for (int v = 0; v < 8; ++v) {
                    float h = fast_tanh(acc[tn][v]);
                    acc[tn][v] = h;
                    sp[kk+1][tn][v >> 1][v & 1] = (_Float16)(1.f - h*h);
                    if (kk < 2) st[(v + lhalf*8)*128 + Nn] = (_Float16)h;
                }
            }
        }

        // ---------------- output scalar: y = h3 . Wout + bout ----------------
        #pragma unroll
        for (int v = 0; v < 8; ++v) {
            float py = 0.f;
            #pragma unroll
            for (int tn = 0; tn < 8; ++tn) py += acc[tn][v] * WoutS[tn*16 + lcol];
            py += __shfl_xor(py, 1); py += __shfl_xor(py, 2);
            py += __shfl_xor(py, 4); py += __shfl_xor(py, 8);
            if (lcol == 0) y_out[r0 + lhalf*8 + v] = py + bout0;
        }

        // ---------------- backward: u3 = Wout * s3 ----------------
        v8f u[8];
        #pragma unroll
        for (int tn = 0; tn < 8; ++tn) {
            float wv = WoutS[tn*16 + lcol];
            #pragma unroll
            for (int v = 0; v < 8; ++v)
                u[tn][v] = wv * (float)sp[3][tn][v >> 1][v & 1];
        }
        #pragma unroll
        for (int kk = 2; kk >= 0; --kk) {           // g_{kk} = u_{kk+1} . Wh[kk]^T
            #pragma unroll
            for (int tn = 0; tn < 8; ++tn) {        // stage u in A-source layout
                int Nn = tn*16 + lcol;
                #pragma unroll
                for (int v = 0; v < 8; ++v)
                    st[(v + lhalf*8)*128 + Nn] = (_Float16)u[tn][v];
            }
            v16h afr[4];
            #pragma unroll
            for (int kf = 0; kf < 4; ++kf) {
                int base = lcol*128 + kf*32 + lhalf*16;
                v8h lo = *(const v8h*)(st + base);
                v8h hi = *(const v8h*)(st + base + 8);
                #pragma unroll
                for (int e = 0; e < 8; ++e) { afr[kf][e] = lo[e]; afr[kf][8+e] = hi[e]; }
            }
            #pragma unroll
            for (int tn = 0; tn < 8; ++tn) {
                int Nn = tn*16 + lcol;
                v8f c;
                #pragma unroll
                for (int v = 0; v < 8; ++v) c[v] = 0.f;
                #pragma unroll
                for (int kf = 0; kf < 4; ++kf) {
                    v16h bfr = *(const v16h*)(Wb + kk*16384 + Nn*128 + kf*32 + lhalf*16);
                    c = __builtin_amdgcn_wmma_f32_16x16x32_f16(
                            false, afr[kf], false, bfr, (short)0, c, false, false);
                }
                #pragma unroll
                for (int v = 0; v < 8; ++v)
                    u[tn][v] = c[v] * (float)sp[kk][tn][v >> 1][v & 1];
            }
        }

        // ---------------- gx[j] = sum_N u0[N] * Win[j][N] ----------------
        #pragma unroll
        for (int v = 0; v < 8; ++v) {
            #pragma unroll
            for (int j = 0; j < 4; ++j) {
                float p = 0.f;
                #pragma unroll
                for (int tn = 0; tn < 8; ++tn)
                    p += u[tn][v] * (float)WinS[j*128 + tn*16 + lcol];
                p += __shfl_xor(p, 1); p += __shfl_xor(p, 2);
                p += __shfl_xor(p, 4); p += __shfl_xor(p, 8);
                if (lcol == 0) g_out[(r0 + lhalf*8 + v)*4 + j] = p;
            }
        }
    }
}

// ---------------------------------------------------------------------------
// Kernel 3: quadrature combine + chain rule + envelope + analytic term
// ---------------------------------------------------------------------------
__global__ void final_kernel(const float4* __restrict__ tx,
                             const float* __restrict__ Yic, const float* __restrict__ Gic,
                             const float* __restrict__ Ydp, const float* __restrict__ Gdp,
                             float* __restrict__ out, int N)
{
    int i = blockIdx.x * blockDim.x + threadIdx.x;
    if (i >= N) return;
    float4 row = tx[i];
    float t = row.x, x = row.y, y = row.z, z = row.w;
    float r2 = x*x + y*y + z*z;
    float r  = sqrtf(r2 + 1e-12f);
    float a  = 0.5f * t;

    const float wq[3] = {5.f/9.f, 8.f/9.f, 5.f/9.f};
    float S = 0.f, Gs0 = Gic[i*4+0], Gs1 = Gic[i*4+1], Gs2 = Gic[i*4+2];
    #pragma unroll
    for (int q = 0; q < 3; ++q) {
        int e = 3*i + q;
        S   += wq[q] * Ydp[e];
        Gs0 += a * wq[q] * Gdp[e*4 + 1];
        Gs1 += a * wq[q] * Gdp[e*4 + 2];
        Gs2 += a * wq[q] * Gdp[e*4 + 3];
    }
    float total = Yic[i] + a * S;

    float env  = -1.f / (1.f + r);
    float denv = 1.f / ((1.f + r) * (1.f + r));     // d(env)/dr

    float invr = 1.f / r;
    float drdx = x*invr, drdy = y*invr, drdz = z*invr;
    float cth  = fminf(fmaxf(z*invr, -1.f), 1.f);
    float s2   = fmaxf(1.f - cth*cth, 1e-12f);
    float dthdc = -1.f / sqrtf(s2);
    float invr3 = invr*invr*invr;
    float dthdx = dthdc * (-z*x*invr3);
    float dthdy = dthdc * (-z*y*invr3);
    float dthdz = dthdc * (invr - z*z*invr3);
    float dxy   = x*x + y*y + 1e-20f;
    float dphdx = -y / dxy, dphdy = x / dxy;

    float inv_as = rsqrtf(r2 + 1e-12f + SOFTENING);
    float ia3 = inv_as * inv_as * inv_as;

    float pot = total * env - inv_as;

    float dpx = env*(Gs0*drdx + Gs1*dthdx + Gs2*dphdx) + total*denv*drdx + x*ia3;
    float dpy = env*(Gs0*drdy + Gs1*dthdy + Gs2*dphdy) + total*denv*drdy + y*ia3;
    float dpz = env*(Gs0*drdz + Gs1*dthdz)             + total*denv*drdz + z*ia3;

    out[i] = pot;
    out[N + i*3 + 0] = -dpx;
    out[N + i*3 + 1] = -dpy;
    out[N + i*3 + 2] = -dpz;
}

// ---------------------------------------------------------------------------
extern "C" void kernel_launch(void* const* d_in, const int* in_sizes, int n_in,
                              void* d_out, int out_size, void* d_ws, size_t ws_size,
                              hipStream_t stream)
{
    const int N = in_sizes[0] / 4;
    const float4* tx      = (const float4*)d_in[0];
    const float*  dp_Win  = (const float*)d_in[1];
    const float*  dp_bin  = (const float*)d_in[2];
    const float*  dp_Wh   = (const float*)d_in[3];
    const float*  dp_bh   = (const float*)d_in[4];
    const float*  dp_Wout = (const float*)d_in[5];
    const float*  dp_bout = (const float*)d_in[6];
    const float*  ic_Win  = (const float*)d_in[7];
    const float*  ic_bin  = (const float*)d_in[8];
    const float*  ic_Wh   = (const float*)d_in[9];
    const float*  ic_bh   = (const float*)d_in[10];
    const float*  ic_Wout = (const float*)d_in[11];
    const float*  ic_bout = (const float*)d_in[12];

    float* ws = (float*)d_ws;
    size_t n = (size_t)N;
    float* ic_in = ws;              // 4N
    float* dp_in = ws + 4*n;        // 12N
    float* Yic   = ws + 16*n;       // N
    float* Ydp   = ws + 17*n;       // 3N
    float* Gic   = ws + 20*n;       // 4N
    float* Gdp   = ws + 24*n;       // 12N  (total 36N floats)

    prep_kernel<<<(N + 255)/256, 256, 0, stream>>>(tx, ic_in, dp_in, N);

    hipFuncSetAttribute((const void*)mlp_eval_kernel,
                        hipFuncAttributeMaxDynamicSharedMemorySize, (int)SMEM_BYTES);

    // delta_phi net: 3N eval rows, Din = 4
    mlp_eval_kernel<<<1024, 256, SMEM_BYTES, stream>>>(
        dp_in, 3*N, 4, dp_Win, dp_bin, dp_Wh, dp_bh, dp_Wout, dp_bout, Ydp, Gdp);
    // initial-correction net: N eval rows, Din = 3
    mlp_eval_kernel<<<512, 256, SMEM_BYTES, stream>>>(
        ic_in, N, 3, ic_Win, ic_bin, ic_Wh, ic_bh, ic_Wout, ic_bout, Yic, Gic);

    final_kernel<<<(N + 255)/256, 256, 0, stream>>>(
        tx, Yic, Gic, Ydp, Gdp, (float*)d_out, N);
}
